// MultiHeadAttention_49855980372570
// MI455X (gfx1250) — compile-verified, matrix operand order checked by reading, not run on hardware
//
#include <hip/hip_runtime.h>

// ---------------------------------------------------------------------------
// MHA for MI455X (gfx1250): bf16 WMMA (v_wmma_f32_16x16x32_bf16) everywhere.
// B=4, T=2048, C=1024, H=16, D=64, INNER=1024, OUT=1024
// Pipeline: (0) cvt f32->bf16 for x and weights, (1) QKV GEMM, (2) flash
// attention (V kept transposed), (3) output GEMM.
// ---------------------------------------------------------------------------

typedef __attribute__((ext_vector_type(16))) __bf16 v16bf;
typedef __attribute__((ext_vector_type(8)))  __bf16 v8bf;
typedef __attribute__((ext_vector_type(4)))  __bf16 v4bf;
typedef __attribute__((ext_vector_type(8)))  float  v8f;

#define B_   4
#define T_   2048
#define C_   1024
#define H_   16
#define D_   64
#define IN_  1024
#define OUT_ 1024

// ---- WMMA wrapper ----------------------------------------------------------
__device__ __forceinline__ v8f wmma_bf16(v16bf a, v16bf b, v8f c) {
    // (neg_a, A, neg_b, B, c_mod, C, reuse_a, reuse_b)
    return __builtin_amdgcn_wmma_f32_16x16x32_bf16(false, a, false, b,
                                                   (short)0, c, false, false);
}

// ---- Fragment loader -------------------------------------------------------
// A 16x32 and B 32x16 bf16 fragments share the same per-lane pattern when the
// source is row-major with K contiguous:
//   lane L: row = (L&15); chunk0 at k0 + (L>>4)*8, chunk1 at k0 + 16 + (L>>4)*8
__device__ __forceinline__ v16bf frag_from_bf16(const __bf16* __restrict__ row,
                                                int k0, int half) {
    v8bf lo = *(const v8bf*)(row + k0 + half * 8);
    v8bf hi = *(const v8bf*)(row + k0 + 16 + half * 8);
    v16bf f;
#pragma unroll
    for (int i = 0; i < 8; ++i) { f[i] = lo[i]; f[8 + i] = hi[i]; }
    return f;
}

// ===========================================================================
// Kernel 0: one-time f32 -> bf16 conversion (4 elems / thread, vectorized)
// ===========================================================================
__global__ __launch_bounds__(256)
void cvt_f32_to_bf16_kernel(const float* __restrict__ src,
                            __bf16* __restrict__ dst, int n4) {
    int i = blockIdx.x * 256 + threadIdx.x;
    if (i < n4) {
        float4 v = ((const float4*)src)[i];
        v4bf o;
        o[0] = (__bf16)v.x; o[1] = (__bf16)v.y;
        o[2] = (__bf16)v.z; o[3] = (__bf16)v.w;
        ((v4bf*)dst)[i] = o;
    }
}

// ===========================================================================
// Kernel 1: QKV projection.  y = (x @ W^T) * C^{-1/2}
// grid = (IN/128, B*T/64, 3), block = 128 (4 waves, each 16 rows x 128 cols)
// which=0 -> Q [B,H,T,D], which=1 -> K [B,H,T,D], which=2 -> V^T [B,H,D,T]
// ===========================================================================
__global__ __launch_bounds__(128)
void qkv_proj_kernel(const __bf16* __restrict__ xb,
                     const __bf16* __restrict__ Wkb,
                     const __bf16* __restrict__ Wqb,
                     const __bf16* __restrict__ Wvb,
                     __bf16* __restrict__ Q,
                     __bf16* __restrict__ K,
                     __bf16* __restrict__ Vt) {
    const int lane = threadIdx.x & 31;
    const int wave = threadIdx.x >> 5;
    const int half = lane >> 4;
    const int lr   = lane & 15;
    const int m0 = (blockIdx.y * 4 + wave) * 16;
    const int n0 = blockIdx.x * 128;
    const int which = blockIdx.z;
    const __bf16* W = (which == 0) ? Wqb : (which == 1) ? Wkb : Wvb;

    const __bf16* xrow = xb + (size_t)(m0 + lr) * C_;
    const __bf16* wrow[8];
#pragma unroll
    for (int j = 0; j < 8; ++j)
        wrow[j] = W + (size_t)(n0 + j * 16 + lr) * C_;

    v8f acc[8];
#pragma unroll
    for (int j = 0; j < 8; ++j)
#pragma unroll
        for (int r = 0; r < 8; ++r) acc[j][r] = 0.0f;

    for (int k0 = 0; k0 < C_; k0 += 32) {
        __builtin_prefetch(xrow + k0 + 128, 0, 1);
        v16bf a = frag_from_bf16(xrow, k0, half);
#pragma unroll
        for (int j = 0; j < 8; ++j) {
            v16bf b = frag_from_bf16(wrow[j], k0, half);
            acc[j] = wmma_bf16(a, b, acc[j]);
        }
    }

    // scale = C^{-1/2} = 1/32
#pragma unroll
    for (int j = 0; j < 8; ++j) {
#pragma unroll
        for (int r = 0; r < 8; ++r) {
            const int m = m0 + r + 8 * half;      // global bt row
            const int i = n0 + j * 16 + lr;       // inner index
            const __bf16 v = (__bf16)(acc[j][r] * 0.03125f);
            const int b = m >> 11, t = m & (T_ - 1);
            const int hh = i >> 6, dd = i & 63;
            if (which == 2)
                Vt[(((size_t)(b * H_ + hh)) * D_ + dd) * T_ + t] = v;
            else if (which == 0)
                Q[(((size_t)(b * H_ + hh)) * T_ + t) * D_ + dd] = v;
            else
                K[(((size_t)(b * H_ + hh)) * T_ + t) * D_ + dd] = v;
        }
    }
}

// ===========================================================================
// Kernel 2: flash attention.  one wave = 32 queries of one (b,h): two q-tiles
// sharing the K/V fragments.  grid = (T/32, H, B), block = 32.
// scores = Q K^T * D^{-1/2}; online softmax; acc += P V  (V stored transposed)
// ===========================================================================
__global__ __launch_bounds__(32)
void attn_kernel(const __bf16* __restrict__ Q,
                 const __bf16* __restrict__ K,
                 const __bf16* __restrict__ Vt,
                 __bf16* __restrict__ O) {
    __shared__ __align__(16) __bf16 Pld[2][16 * 32];

    const int lane = threadIdx.x & 31;
    const int half = lane >> 4;
    const int lr   = lane & 15;
    const int q0 = blockIdx.x * 32;
    const int h  = blockIdx.y;
    const int b  = blockIdx.z;

    const __bf16* Qb = Q  + ((size_t)(b * H_ + h)) * T_ * D_;
    const __bf16* Kb = K  + ((size_t)(b * H_ + h)) * T_ * D_;
    const __bf16* Vb = Vt + ((size_t)(b * H_ + h)) * D_ * T_;

    v16bf aq[2][2];
#pragma unroll
    for (int g = 0; g < 2; ++g) {
        const __bf16* qrow = Qb + (size_t)(q0 + g * 16 + lr) * D_;
        aq[g][0] = frag_from_bf16(qrow, 0, half);   // d = 0..31
        aq[g][1] = frag_from_bf16(qrow, 32, half);  // d = 32..63
    }

    float mrow[2][8], lrow[2][8];
    v8f acc[2][4];
#pragma unroll
    for (int g = 0; g < 2; ++g) {
#pragma unroll
        for (int r = 0; r < 8; ++r) { mrow[g][r] = -3.0e38f; lrow[g][r] = 0.0f; }
#pragma unroll
        for (int j = 0; j < 4; ++j)
#pragma unroll
            for (int r = 0; r < 8; ++r) acc[g][j][r] = 0.0f;
    }

    for (int kt = 0; kt < T_; kt += 32) {
        // ---- K fragments for 32 keys (shared by both q-tiles) ----
        const __bf16* kr0 = Kb + (size_t)(kt + lr) * D_;
        const __bf16* kr1 = Kb + (size_t)(kt + 16 + lr) * D_;
        const v16bf kb0_0 = frag_from_bf16(kr0, 0, half);
        const v16bf kb0_1 = frag_from_bf16(kr0, 32, half);
        const v16bf kb1_0 = frag_from_bf16(kr1, 0, half);
        const v16bf kb1_1 = frag_from_bf16(kr1, 32, half);

#pragma unroll
        for (int g = 0; g < 2; ++g) {
            v8f s0, s1;
#pragma unroll
            for (int r = 0; r < 8; ++r) { s0[r] = 0.0f; s1[r] = 0.0f; }
            s0 = wmma_bf16(aq[g][0], kb0_0, s0);
            s0 = wmma_bf16(aq[g][1], kb0_1, s0);
            s1 = wmma_bf16(aq[g][0], kb1_0, s1);
            s1 = wmma_bf16(aq[g][1], kb1_1, s1);

            // ---- online softmax (row m = vgpr r + 8*half, cols across lanes)
            float alpha[8];
#pragma unroll
            for (int r = 0; r < 8; ++r) {
                float x0 = s0[r] * 0.125f;   // D^{-1/2}
                float x1 = s1[r] * 0.125f;
                float t = fmaxf(x0, x1);
                t = fmaxf(t, __shfl_xor(t, 1));
                t = fmaxf(t, __shfl_xor(t, 2));
                t = fmaxf(t, __shfl_xor(t, 4));
                t = fmaxf(t, __shfl_xor(t, 8));
                const float mn = fmaxf(mrow[g][r], t);
                const float al = __expf(mrow[g][r] - mn);
                const float p0 = __expf(x0 - mn);
                const float p1 = __expf(x1 - mn);
                float rs = p0 + p1;
                rs += __shfl_xor(rs, 1);
                rs += __shfl_xor(rs, 2);
                rs += __shfl_xor(rs, 4);
                rs += __shfl_xor(rs, 8);
                lrow[g][r] = lrow[g][r] * al + rs;
                mrow[g][r] = mn;
                alpha[r] = al;
                // C-layout -> row-major 16x32 bf16 P tile in LDS
                Pld[g][(r + 8 * half) * 32 + lr]      = (__bf16)p0;
                Pld[g][(r + 8 * half) * 32 + 16 + lr] = (__bf16)p1;
            }
#pragma unroll
            for (int j = 0; j < 4; ++j)
#pragma unroll
                for (int r = 0; r < 8; ++r) acc[g][j][r] *= alpha[r];
        }

        __syncthreads();
        const v16bf pa0 = frag_from_bf16(&Pld[0][lr * 32], 0, half);
        const v16bf pa1 = frag_from_bf16(&Pld[1][lr * 32], 0, half);
#pragma unroll
        for (int j = 0; j < 4; ++j) {
            const __bf16* vrow = Vb + (size_t)(j * 16 + lr) * T_; // V^T row = d
            const v16bf bv = frag_from_bf16(vrow, kt, half);
            acc[0][j] = wmma_bf16(pa0, bv, acc[0][j]);
            acc[1][j] = wmma_bf16(pa1, bv, acc[1][j]);
        }
        __syncthreads();
    }

    // ---- normalize + store O as bf16 [B*T, INNER] ----
#pragma unroll
    for (int g = 0; g < 2; ++g) {
        float inv[8];
#pragma unroll
        for (int r = 0; r < 8; ++r) inv[r] = 1.0f / lrow[g][r];
#pragma unroll
        for (int j = 0; j < 4; ++j) {
#pragma unroll
            for (int r = 0; r < 8; ++r) {
                const int q = q0 + g * 16 + r + 8 * half;
                const int i = h * D_ + j * 16 + lr;
                O[((size_t)(b * T_ + q)) * IN_ + i] =
                    (__bf16)(acc[g][j][r] * inv[r]);
            }
        }
    }
}

// ===========================================================================
// Kernel 3: output projection.  out = (O @ Wp^T) * INNER^{-1/2}, f32 out.
// grid = (OUT/128, B*T/64), block = 128 (4 waves, each 16 rows x 128 cols)
// ===========================================================================
__global__ __launch_bounds__(128)
void out_proj_kernel(const __bf16* __restrict__ O,
                     const __bf16* __restrict__ Wpb,
                     float* __restrict__ out) {
    const int lane = threadIdx.x & 31;
    const int wave = threadIdx.x >> 5;
    const int half = lane >> 4;
    const int lr   = lane & 15;
    const int m0 = (blockIdx.y * 4 + wave) * 16;
    const int n0 = blockIdx.x * 128;

    const __bf16* orow = O + (size_t)(m0 + lr) * IN_;
    const __bf16* wrow[8];
#pragma unroll
    for (int j = 0; j < 8; ++j)
        wrow[j] = Wpb + (size_t)(n0 + j * 16 + lr) * IN_;

    v8f acc[8];
#pragma unroll
    for (int j = 0; j < 8; ++j)
#pragma unroll
        for (int r = 0; r < 8; ++r) acc[j][r] = 0.0f;

    for (int k0 = 0; k0 < IN_; k0 += 32) {
        __builtin_prefetch(orow + k0 + 128, 0, 1);
        v16bf a = frag_from_bf16(orow, k0, half);
#pragma unroll
        for (int j = 0; j < 8; ++j) {
            v16bf b = frag_from_bf16(wrow[j], k0, half);
            acc[j] = wmma_bf16(a, b, acc[j]);
        }
    }

    // scale = INNER^{-1/2} = 1/32
#pragma unroll
    for (int j = 0; j < 8; ++j) {
#pragma unroll
        for (int r = 0; r < 8; ++r) {
            const int m = m0 + r + 8 * half;
            const int n = n0 + j * 16 + lr;
            out[(size_t)m * OUT_ + n] = acc[j][r] * 0.03125f;
        }
    }
}

// ===========================================================================
extern "C" void kernel_launch(void* const* d_in, const int* in_sizes, int n_in,
                              void* d_out, int out_size, void* d_ws, size_t ws_size,
                              hipStream_t stream) {
    const float* x  = (const float*)d_in[0];
    const float* Wk = (const float*)d_in[1];
    const float* Wq = (const float*)d_in[2];
    const float* Wv = (const float*)d_in[3];
    const float* Wp = (const float*)d_in[4];
    float* out = (float*)d_out;

    // ---- workspace layout (bytes) ----
    // xb 16MB | Wkb 2MB | Wqb 2MB | Wvb 2MB | Wpb 2MB | Q 16MB | K 16MB
    // | Vt 16MB | O 16MB   (total 88MB)
    char* ws = (char*)d_ws;
    const size_t xBytes = (size_t)B_ * T_ * C_ * 2;   // 16 MB
    const size_t wBytes = (size_t)IN_ * C_ * 2;       // 2 MB
    const size_t sBytes = (size_t)B_ * H_ * T_ * D_ * 2; // 16 MB
    __bf16* xb  = (__bf16*)(ws);
    __bf16* Wkb = (__bf16*)(ws + xBytes);
    __bf16* Wqb = (__bf16*)(ws + xBytes + 1 * wBytes);
    __bf16* Wvb = (__bf16*)(ws + xBytes + 2 * wBytes);
    __bf16* Wpb = (__bf16*)(ws + xBytes + 3 * wBytes);
    __bf16* Q   = (__bf16*)(ws + xBytes + 4 * wBytes);
    __bf16* K   = (__bf16*)(ws + xBytes + 4 * wBytes + 1 * sBytes);
    __bf16* Vt  = (__bf16*)(ws + xBytes + 4 * wBytes + 2 * sBytes);
    __bf16* O   = (__bf16*)(ws + xBytes + 4 * wBytes + 3 * sBytes);

    // ---- phase 0: one-time bf16 conversions ----
    const int xN4 = (B_ * T_ * C_) / 4;   // 2097152
    const int wN4 = (IN_ * C_) / 4;       // 262144
    cvt_f32_to_bf16_kernel<<<xN4 / 256, 256, 0, stream>>>(x, xb, xN4);
    cvt_f32_to_bf16_kernel<<<wN4 / 256, 256, 0, stream>>>(Wk, Wkb, wN4);
    cvt_f32_to_bf16_kernel<<<wN4 / 256, 256, 0, stream>>>(Wq, Wqb, wN4);
    cvt_f32_to_bf16_kernel<<<wN4 / 256, 256, 0, stream>>>(Wv, Wvb, wN4);
    cvt_f32_to_bf16_kernel<<<wN4 / 256, 256, 0, stream>>>(Wp, Wpb, wN4);

    // ---- phase 1: QKV projections ----
    dim3 g1(IN_ / 128, (B_ * T_) / 64, 3);
    qkv_proj_kernel<<<g1, 128, 0, stream>>>(xb, Wkb, Wqb, Wvb, Q, K, Vt);

    // ---- phase 2: flash attention ----
    dim3 g2(T_ / 32, H_, B_);
    attn_kernel<<<g2, 32, 0, stream>>>(Q, K, Vt, O);

    // ---- phase 3: output projection ----
    dim3 g3(OUT_ / 128, (B_ * T_) / 64, 1);
    out_proj_kernel<<<g3, 128, 0, stream>>>(O, Wpb, out);
}
